// CrossFormerBlocks_18425409700030
// MI455X (gfx1250) — matrix operand order repaired and checked by reading, not compiled, not run on hardware
//
#include <hip/hip_runtime.h>
#include <stdint.h>

// ---------------------------------------------------------------------------
// CrossFormer blocks for MI455X (gfx1250, wave32, WMMA + TDM).
// All GEMMs (Q/K/V/O projections, 9-tap conv-as-GEMM, FFN linear) and the
// attention score / PV products run on v_wmma_f32_16x16x32_bf16.
// A-tiles of the dense GEMMs are DMA'd into LDS by the Tensor Data Mover
// (tensor_load_to_lds + s_wait_tensorcnt); fragments are preloaded so one
// s_wait_dscnt covers a burst of independent WMMAs.
// ---------------------------------------------------------------------------

typedef __attribute__((ext_vector_type(16))) __bf16 v16bf;
typedef __attribute__((ext_vector_type(8)))  float  v8f;
typedef __attribute__((ext_vector_type(4)))  unsigned int v4u;
typedef __attribute__((ext_vector_type(8)))  int v8i;
typedef __attribute__((ext_vector_type(4)))  int v4i;

union FragB { v16bf v; uint4 q[2]; };

#define T1N 2048
#define T2N 512
#define CN  384
#define DN  192

#if __has_builtin(__builtin_amdgcn_tensor_load_to_lds) && \
    __has_builtin(__builtin_amdgcn_s_wait_tensorcnt)
#define HAVE_TDM 1
#else
#define HAVE_TDM 0
#endif

__device__ __forceinline__ __bf16 f2bf(float x) {
  union { float f; unsigned u; } c; c.f = x;
  unsigned r = (c.u + 0x7FFFu + ((c.u >> 16) & 1u)) >> 16;
  union { unsigned short s; __bf16 b; } o; o.s = (unsigned short)r;
  return o.b;
}

__device__ __forceinline__ v8f wmma_bf16(const FragB& a, const FragB& b, v8f c) {
  return __builtin_amdgcn_wmma_f32_16x16x32_bf16(false, a.v, false, b.v,
                                                 (short)0, c, false, false);
}

#if HAVE_TDM
// Issue a TDM load of a 2D bf16 tile [tile_rows x 32 elems] from a row-major
// tensor with row stride `stride_e` elements into LDS at byte offset lds_off,
// padding the LDS destination by 4 DWORDs after every 16 DWORDs (i.e. LDS row
// stride 40 halfs for a 32-half tile row).
__device__ __forceinline__ void tdm_load_tile_a(const __bf16* gaddr, unsigned lds_off,
                                                unsigned stride_e)
{
  unsigned long long ga = (unsigned long long)(uintptr_t)gaddr;
  v4u g0 = {};
  g0[0] = 1u;                                   // count=1 (valid user D#)
  g0[1] = lds_off;                              // lds_addr (bytes)
  g0[2] = (unsigned)(ga & 0xFFFFFFFFu);         // global_addr[31:0]
  g0[3] = (unsigned)((ga >> 32) & 0x01FFFFFFu)  // global_addr[56:32]
        | (2u << 30);                           // type = 2 ("image")
  v8i g1 = {};
  unsigned dw0 = (1u << 16)                     // data_size = 2 bytes
               | (1u << 20)                     // pad_enable
               | (3u << 22)                     // pad_interval: 16 DWORDs
               | (3u << 25);                    // pad_amount: 4 DWORDs
  const unsigned tdim0 = stride_e;              // tensor dim0 (elements)
  const unsigned tdim1 = 0x40000000u;           // effectively unbounded rows
  g1[0] = (int)dw0;
  g1[1] = (int)((tdim0 & 0xFFFFu) << 16);                       // dim0 lo16
  g1[2] = (int)(((tdim0 >> 16) & 0xFFFFu) | ((tdim1 & 0xFFFFu) << 16));
  g1[3] = (int)(((tdim1 >> 16) & 0xFFFFu) | (32u << 16));       // tile_dim0=32
  g1[4] = (int)128u;                                            // tile_dim1=128
  g1[5] = (int)stride_e;                                        // dim0 stride lo32
  g1[6] = 0;
  g1[7] = 0;
  v4i z4 = {};
#if defined(__clang_major__) && __clang_major__ >= 23
  v8i z8 = {};
  __builtin_amdgcn_tensor_load_to_lds(g0, g1, z4, z4, z8, 0);
#else
  __builtin_amdgcn_tensor_load_to_lds(g0, g1, z4, z4, 0);
#endif
}
#endif

// ---------------------------------------------------------------------------
// Generic WMMA GEMM:  out[M,N] = A[M,Kd] (bf16 row-major) x W[(TAPS,)Kd,N]
// Block: 256 threads / 8 waves; tile 128(M) x 64(N) x 32(K); wave = 16 rows.
// EP 0: out_bf16 = acc * scale
// EP 1: out_f32 += acc + bias (residual add)
// EP 2: out_bf16 = gelu((acc + bias) * scale)      (conv epilogue)
// TAPS==9: k=9 "SAME" conv along time; A rows shifted by tap-4 with halo guard
//          (manual staging).  TAPS==1: A tile staged by the Tensor Data Mover.
// ---------------------------------------------------------------------------
template<int EP, int TAPS>
__global__ __launch_bounds__(256)
void gemm_wmma_kernel(const __bf16* __restrict__ A, int Kd,
                      const __bf16* __restrict__ W, int N,
                      const float* __restrict__ bias, float scale,
                      float* __restrict__ outF, __bf16* __restrict__ outB)
{
  __shared__ __bf16 As[128][40];   // row-major A tile, 32 halfs + 8 pad
  __shared__ __bf16 Bs[64][40];    // B tile transposed: Bs[n][k]

  const int mblk = blockIdx.x * 128;
  const int nblk = blockIdx.y * 64;
  const int tid  = threadIdx.x;
  const int wave = tid >> 5, lane = tid & 31;
  const int lg = lane >> 4, ln = lane & 15;

  v8f acc[4];
  v8f zero8 = {};
#pragma unroll
  for (int t = 0; t < 4; ++t) acc[t] = zero8;

  for (int tap = 0; tap < TAPS; ++tap) {
    const int shift = (TAPS > 1) ? (tap - 4) : 0;
    const __bf16* Wt = W + (size_t)tap * Kd * N;
    for (int k0 = 0; k0 < Kd; k0 += 32) {
      __syncthreads();
      bool manual_a = true;
#if HAVE_TDM
      if (TAPS == 1) {
        manual_a = false;
        if (wave == 0)
          tdm_load_tile_a(A + (size_t)mblk * Kd + k0,
                          (unsigned)(uintptr_t)&As[0][0], (unsigned)Kd);
      }
#endif
      // --- stage B tile transposed: 32 x 64 -> Bs[n][k] (all threads) ---
      {
        int kk = tid >> 3, nn = (tid & 7) * 8;
        uint4 val = *(const uint4*)(Wt + (size_t)(k0 + kk) * N + nblk + nn);
        const __bf16* h = (const __bf16*)&val;
#pragma unroll
        for (int j = 0; j < 8; ++j) Bs[nn + j][kk] = h[j];
      }
      if (manual_a) {
        // --- stage A tile: 128 x 32 halfs = 512 uint4 / 256 threads ---
        for (int i = tid; i < 512; i += 256) {
          int r = i >> 2, cc = (i & 3) * 8;
          int row = mblk + r;
          uint4 val = make_uint4(0u, 0u, 0u, 0u);
          bool ok = true;
          if (TAPS > 1) {
            int t = row & (T1N - 1);
            ok = ((unsigned)(t + shift) < (unsigned)T1N);
          }
          if (ok) val = *(const uint4*)(A + (size_t)(row + shift) * Kd + k0 + cc);
          *(uint4*)&As[r][cc] = val;
        }
      }
      if (k0 + 32 < Kd)
        __builtin_prefetch(Wt + (size_t)(k0 + 32 + (tid >> 3)) * N + nblk, 0, 0);
#if HAVE_TDM
      if (TAPS == 1 && wave == 0) __builtin_amdgcn_s_wait_tensorcnt(0);
#endif
      __syncthreads();

      // Preload A fragment + all 4 B fragments, then burst 4 WMMAs.
      FragB a;
      a.q[0] = *(const uint4*)&As[wave * 16 + ln][8 * lg];
      a.q[1] = *(const uint4*)&As[wave * 16 + ln][8 * lg + 16];
      FragB bf[4];
#pragma unroll
      for (int nt = 0; nt < 4; ++nt) {
        bf[nt].q[0] = *(const uint4*)&Bs[nt * 16 + ln][8 * lg];
        bf[nt].q[1] = *(const uint4*)&Bs[nt * 16 + ln][8 * lg + 16];
      }
#pragma unroll
      for (int nt = 0; nt < 4; ++nt) acc[nt] = wmma_bf16(a, bf[nt], acc[nt]);
    }
  }

  // Epilogue. C/D layout: row = wave*16 + 8*lg + v, col = nblk + nt*16 + ln
#pragma unroll
  for (int nt = 0; nt < 4; ++nt) {
    int col = nblk + nt * 16 + ln;
#pragma unroll
    for (int v = 0; v < 8; ++v) {
      int row = mblk + wave * 16 + 8 * lg + v;
      size_t idx = (size_t)row * N + col;
      float x = acc[nt][v];
      if (EP == 0) {
        outB[idx] = f2bf(x * scale);
      } else if (EP == 1) {
        float bb = bias ? bias[col] : 0.f;
        outF[idx] += x + bb;
      } else {
        float t0 = (x + bias[col]) * scale;
        outB[idx] = f2bf(0.5f * t0 * (1.f + erff(t0 * 0.70710678f)));
      }
    }
  }
}

// ---------------------------------------------------------------------------
// Fused windowed cross-attention, one (b,h) x 128-query tile per block.
// 8 waves x 16 queries; K/V streamed in 64-key chunks; online softmax.
// Layer 0 additionally accumulates the guided-attention loss.
// ---------------------------------------------------------------------------
__global__ __launch_bounds__(256)
void attention_kernel(const __bf16* __restrict__ Q, const __bf16* __restrict__ K,
                      const __bf16* __restrict__ V, __bf16* __restrict__ O,
                      float* __restrict__ lossAcc, int layer0)
{
  __shared__ __bf16 Qs[128][200];      // 128 q x 192 d (+pad)
  __shared__ __bf16 Ks[64][200];       // 64 k x 192 d
  __shared__ __bf16 Vt[192][72];       // transposed: d x 64 k
  __shared__ __bf16 Ps[8][16][72];     // per-wave probs 16 q x 64 k

  const int bh = blockIdx.y;
  const int b = bh >> 1, h = bh & 1;
  const int q0 = blockIdx.x * 128;
  const int tid = threadIdx.x, wave = tid >> 5, lane = tid & 31;
  const int lg = lane >> 4, ln = lane & 15;

  const __bf16* Qg = Q + (size_t)b * T1N * CN + (size_t)h * DN;
  const __bf16* Kg = K + (size_t)b * T2N * CN + (size_t)h * DN;
  const __bf16* Vg = V + (size_t)b * T2N * CN + (size_t)h * DN;

  for (int i = tid; i < 128 * 24; i += 256) {
    int r = i / 24, cc = (i % 24) * 8;
    *(uint4*)&Qs[r][cc] = *(const uint4*)(Qg + (size_t)(q0 + r) * CN + cc);
  }

  float Mrun[8], Lrun[8], Grun[8];
  v8f Ov[12];
  v8f zero8 = {};
#pragma unroll
  for (int v = 0; v < 8; ++v) { Mrun[v] = -1e30f; Lrun[v] = 0.f; Grun[v] = 0.f; }
#pragma unroll
  for (int t = 0; t < 12; ++t) Ov[t] = zero8;

  for (int c = 0; c < 8; ++c) {            // 8 chunks x 64 keys = 512
    const int kbase = c * 64;
    __syncthreads();
    for (int i = tid; i < 64 * 24; i += 256) {
      int r = i / 24, cc = (i % 24) * 8;
      *(uint4*)&Ks[r][cc] = *(const uint4*)(Kg + (size_t)(kbase + r) * CN + cc);
    }
    for (int i = tid; i < 64 * 192; i += 256) {
      int r = i / 192, d = i % 192;
      Vt[d][r] = Vg[(size_t)(kbase + r) * CN + d];
    }
    __syncthreads();

    // ---- scores S[16 q x 64 k] = Q_w (16x192) x K^T ----
    v8f S[4];
#pragma unroll
    for (int nt = 0; nt < 4; ++nt) S[nt] = zero8;
#pragma unroll
    for (int d0 = 0; d0 < 192; d0 += 32) {
      FragB a;
      a.q[0] = *(const uint4*)&Qs[wave * 16 + ln][d0 + 8 * lg];
      a.q[1] = *(const uint4*)&Qs[wave * 16 + ln][d0 + 8 * lg + 16];
      FragB kb[4];
#pragma unroll
      for (int nt = 0; nt < 4; ++nt) {
        kb[nt].q[0] = *(const uint4*)&Ks[nt * 16 + ln][d0 + 8 * lg];
        kb[nt].q[1] = *(const uint4*)&Ks[nt * 16 + ln][d0 + 8 * lg + 16];
      }
#pragma unroll
      for (int nt = 0; nt < 4; ++nt) S[nt] = wmma_bf16(a, kb[nt], S[nt]);
    }

    // ---- sliding-window mask + chunk row max ----
    float cmax[8];
#pragma unroll
    for (int v = 0; v < 8; ++v) cmax[v] = -1e30f;
#pragma unroll
    for (int nt = 0; nt < 4; ++nt) {
      int key = kbase + nt * 16 + ln;
#pragma unroll
      for (int v = 0; v < 8; ++v) {
        int qrow = q0 + wave * 16 + 8 * lg + v;
        int ctr = qrow >> 2;                       // (q*T2)/T1
        int lo = (ctr - 51 > 0) ? ctr - 51 : 0;    // w = 51
        bool ok = (key >= lo) && (key < ctr + 51);
        float s = ok ? S[nt][v] : -1e9f;
        S[nt][v] = s;
        cmax[v] = fmaxf(cmax[v], s);
      }
    }
#pragma unroll
    for (int v = 0; v < 8; ++v) {
      float m = cmax[v];
#pragma unroll
      for (int off = 1; off < 16; off <<= 1) m = fmaxf(m, __shfl_xor(m, off));
      cmax[v] = m;
    }

    float alpha[8], ls[8], gs[8];
#pragma unroll
    for (int v = 0; v < 8; ++v) {
      float nm = fmaxf(Mrun[v], cmax[v]);
      alpha[v] = __expf(Mrun[v] - nm);
      Mrun[v] = nm;
      ls[v] = 0.f; gs[v] = 0.f;
    }
#pragma unroll
    for (int t = 0; t < 12; ++t)
#pragma unroll
      for (int v = 0; v < 8; ++v) Ov[t][v] *= alpha[v];

    // ---- probs, row sums, guided-mask sums; spill P to LDS for PV GEMM ----
#pragma unroll
    for (int nt = 0; nt < 4; ++nt) {
      int key = kbase + nt * 16 + ln;
      float sj = key * (1.f / 512.f);
#pragma unroll
      for (int v = 0; v < 8; ++v) {
        int qrow = q0 + wave * 16 + 8 * lg + v;
        float pe = __expf(S[nt][v] - Mrun[v]);
        float ti = qrow * (1.f / 2048.f);
        float df = sj - ti;
        float gm = 1.f - __expf(df * df * (-1.f / 0.18f));  // 2*sigma^2 = 0.18
        ls[v] += pe;
        gs[v] += pe * gm;
        Ps[wave][8 * lg + v][nt * 16 + ln] = f2bf(pe);
      }
    }
#pragma unroll
    for (int v = 0; v < 8; ++v) {
      float l = ls[v], g = gs[v];
#pragma unroll
      for (int off = 1; off < 16; off <<= 1) {
        l += __shfl_xor(l, off);
        g += __shfl_xor(g, off);
      }
      Lrun[v] = Lrun[v] * alpha[v] + l;
      Grun[v] = Grun[v] * alpha[v] + g;
    }

    // ---- O += P (16x64) x V (64x192), two d-tiles per burst ----
    FragB pa[2];
#pragma unroll
    for (int kk = 0; kk < 2; ++kk) {
      pa[kk].q[0] = *(const uint4*)&Ps[wave][ln][kk * 32 + 8 * lg];
      pa[kk].q[1] = *(const uint4*)&Ps[wave][ln][kk * 32 + 8 * lg + 16];
    }
#pragma unroll
    for (int dt = 0; dt < 12; dt += 2) {
      FragB vb[4];
#pragma unroll
      for (int u = 0; u < 2; ++u)
#pragma unroll
        for (int kk = 0; kk < 2; ++kk) {
          vb[u * 2 + kk].q[0] = *(const uint4*)&Vt[(dt + u) * 16 + ln][kk * 32 + 8 * lg];
          vb[u * 2 + kk].q[1] = *(const uint4*)&Vt[(dt + u) * 16 + ln][kk * 32 + 8 * lg + 16];
        }
      Ov[dt]     = wmma_bf16(pa[0], vb[0], Ov[dt]);
      Ov[dt]     = wmma_bf16(pa[1], vb[1], Ov[dt]);
      Ov[dt + 1] = wmma_bf16(pa[0], vb[2], Ov[dt + 1]);
      Ov[dt + 1] = wmma_bf16(pa[1], vb[3], Ov[dt + 1]);
    }
  }

  // ---- normalize + store (concat heads: col = h*192 + d) ----
  __bf16* Og = O + (size_t)b * T1N * CN + (size_t)h * DN;
  float rl[8];
#pragma unroll
  for (int v = 0; v < 8; ++v) rl[v] = 1.f / Lrun[v];
#pragma unroll
  for (int dt = 0; dt < 12; ++dt) {
    int d = dt * 16 + ln;
#pragma unroll
    for (int v = 0; v < 8; ++v) {
      int qrow = q0 + wave * 16 + 8 * lg + v;
      Og[(size_t)qrow * CN + d] = f2bf(Ov[dt][v] * rl[v]);
    }
  }
  if (layer0 && ln == 0) {
    float gsum = 0.f;
#pragma unroll
    for (int v = 0; v < 8; ++v) gsum += Grun[v] * rl[v];
    atomicAdd(lossAcc, gsum);
  }
}

// ---------------------------------------------------------------------------
// LayerNorm over rows of 384; block = 128 threads, 3 elements/thread.
// outB != null -> bf16 output; else f32 output (final LN -> d_out).
// ---------------------------------------------------------------------------
__global__ __launch_bounds__(128)
void layernorm_kernel(const float* __restrict__ x, const float* __restrict__ g,
                      const float* __restrict__ bt, __bf16* __restrict__ outB,
                      float* __restrict__ outF)
{
  const int row = blockIdx.x, tid = threadIdx.x;
  const float* xr = x + (size_t)row * CN;
  float v0 = xr[tid], v1 = xr[tid + 128], v2 = xr[tid + 256];
  __shared__ float red[128];
  red[tid] = v0 + v1 + v2;
  __syncthreads();
  for (int o = 64; o > 0; o >>= 1) {
    if (tid < o) red[tid] += red[tid + o];
    __syncthreads();
  }
  float mean = red[0] * (1.f / 384.f);
  __syncthreads();
  float d0 = v0 - mean, d1 = v1 - mean, d2 = v2 - mean;
  red[tid] = d0 * d0 + d1 * d1 + d2 * d2;
  __syncthreads();
  for (int o = 64; o > 0; o >>= 1) {
    if (tid < o) red[tid] += red[tid + o];
    __syncthreads();
  }
  float rstd = rsqrtf(red[0] * (1.f / 384.f) + 1e-5f);
  float y0 = d0 * rstd * g[tid] + bt[tid];
  float y1 = d1 * rstd * g[tid + 128] + bt[tid + 128];
  float y2 = d2 * rstd * g[tid + 256] + bt[tid + 256];
  if (outB) {
    __bf16* op = outB + (size_t)row * CN;
    op[tid] = f2bf(y0); op[tid + 128] = f2bf(y1); op[tid + 256] = f2bf(y2);
  } else {
    float* op = outF + (size_t)row * CN;
    op[tid] = y0; op[tid + 128] = y1; op[tid + 256] = y2;
  }
}

// ---- sinusoidal positional embedding add (fairseq, positions 1..T) ----
__global__ void pos_add_kernel(const float* __restrict__ x, const float* __restrict__ alpha,
                               float* __restrict__ outF, __bf16* __restrict__ outB,
                               int T, int total)
{
  int i = blockIdx.x * blockDim.x + threadIdx.x;
  if (i >= total) return;
  int c = i % CN;
  int t = (i / CN) % T;
  int j = (c < 192) ? c : c - 192;
  float freq = __expf(-0.0482216773f * (float)j);   // ln(1e4)/191
  float ang = (float)(t + 1) * freq;
  float e = (c < 192) ? __sinf(ang) : __cosf(ang);
  float r = x[i] + alpha[0] * e;
  if (outF) outF[i] = r; else outB[i] = f2bf(r);
}

// ---- W[Nout][Kin] (f32) -> Wt[Kin][Nout] (bf16) ----
__global__ void wtrans_kernel(const float* __restrict__ W, __bf16* __restrict__ out,
                              int Nout, int Kin)
{
  int i = blockIdx.x * blockDim.x + threadIdx.x;
  if (i >= Nout * Kin) return;
  int n = i / Kin, k = i % Kin;
  out[(size_t)k * Nout + n] = f2bf(W[i]);
}

// ---- cw[1536][384][9] (f32) -> cwt[9][384][1536] (bf16) ----
__global__ void convw_kernel(const float* __restrict__ W, __bf16* __restrict__ out)
{
  int i = blockIdx.x * blockDim.x + threadIdx.x;
  if (i >= 1536 * 384 * 9) return;
  int f = i / (384 * 9);
  int c = (i / 9) % 384;
  int kk = i % 9;
  out[((size_t)kk * 384 + c) * 1536 + f] = f2bf(W[i]);
}

__global__ void zero_kernel(float* p) {
  if (threadIdx.x == 0 && blockIdx.x == 0) p[0] = 0.f;
}
__global__ void loss_kernel(const float* acc, float* out) {
  if (threadIdx.x == 0 && blockIdx.x == 0)
    out[0] = acc[0] * (1.f / 16777216.f);   // / (H*B*T1*T2)
}

// ---------------------------------------------------------------------------
extern "C" void kernel_launch(void* const* d_in, const int* in_sizes, int n_in,
                              void* d_out, int out_size, void* d_ws, size_t ws_size,
                              hipStream_t stream)
{
  const float* x1    = (const float*)d_in[0];
  const float* x2    = (const float*)d_in[1];
  const float* alpha = (const float*)d_in[2];
  const float* ln1_g = (const float*)d_in[3];
  const float* ln1_b = (const float*)d_in[4];
  const float* qw    = (const float*)d_in[5];
  const float* kw    = (const float*)d_in[6];
  const float* vw    = (const float*)d_in[7];
  const float* ow    = (const float*)d_in[8];
  const float* ln2_g = (const float*)d_in[9];
  const float* ln2_b = (const float*)d_in[10];
  const float* cw    = (const float*)d_in[11];
  const float* cb    = (const float*)d_in[12];
  const float* lw    = (const float*)d_in[13];
  const float* lb    = (const float*)d_in[14];
  const float* lnf_g = (const float*)d_in[15];
  const float* lnf_b = (const float*)d_in[16];
  float* out = (float*)d_out;
  (void)in_sizes; (void)n_in; (void)out_size; (void)ws_size;

  const size_t M1 = 8 * 2048;   // x1 rows
  const size_t M2 = 8 * 512;    // x2 rows

  char* ws = (char*)d_ws;
  size_t off = 0;
  auto take = [&](size_t bytes) -> void* {
    void* p = ws + off;
    off += (bytes + 255) & ~(size_t)255;
    return p;
  };
  float*  x1p = (float*) take(M1 * CN * 4);            // running x1 state (f32)
  __bf16* x2b = (__bf16*)take(M2 * CN * 2);            // x2 + pos (bf16)
  __bf16* qnb = (__bf16*)take(M1 * CN * 2);            // LN1 out / LN2 out
  __bf16* qb  = (__bf16*)take(M1 * CN * 2);
  __bf16* kb  = (__bf16*)take(M2 * CN * 2);
  __bf16* vb  = (__bf16*)take(M2 * CN * 2);
  __bf16* ab  = (__bf16*)take(M1 * CN * 2);            // attention output
  __bf16* hb  = (__bf16*)take(M1 * 1536 * 2);          // gelu(conv) output
  __bf16* qwt = (__bf16*)take(384 * 384 * 2);
  __bf16* kwt = (__bf16*)take(384 * 384 * 2);
  __bf16* vwt = (__bf16*)take(384 * 384 * 2);
  __bf16* owt = (__bf16*)take(384 * 384 * 2);
  __bf16* lwt = (__bf16*)take((size_t)1536 * 384 * 2);
  __bf16* cwt = (__bf16*)take((size_t)9 * 384 * 1536 * 2);
  float*  loss = (float*)take(256);

  zero_kernel<<<1, 32, 0, stream>>>(loss);
  {
    int tot1 = (int)(M1 * CN);
    pos_add_kernel<<<(tot1 + 255) / 256, 256, 0, stream>>>(x1, alpha, x1p, nullptr, 2048, tot1);
    int tot2 = (int)(M2 * CN);
    pos_add_kernel<<<(tot2 + 255) / 256, 256, 0, stream>>>(x2, alpha, nullptr, x2b, 512, tot2);
  }

  const float scaling = 0.07216878364870323f;  // 192^-0.5
  const unsigned g1 = (unsigned)(M1 / 128), g2 = (unsigned)(M2 / 128);

  for (int i = 0; i < 2; ++i) {
    // per-layer bf16 transposed weights
    wtrans_kernel<<<(384 * 384 + 255) / 256, 256, 0, stream>>>(qw + (size_t)i * 384 * 384, qwt, 384, 384);
    wtrans_kernel<<<(384 * 384 + 255) / 256, 256, 0, stream>>>(kw + (size_t)i * 384 * 384, kwt, 384, 384);
    wtrans_kernel<<<(384 * 384 + 255) / 256, 256, 0, stream>>>(vw + (size_t)i * 384 * 384, vwt, 384, 384);
    wtrans_kernel<<<(384 * 384 + 255) / 256, 256, 0, stream>>>(ow + (size_t)i * 384 * 384, owt, 384, 384);
    wtrans_kernel<<<(384 * 1536 + 255) / 256, 256, 0, stream>>>(lw + (size_t)i * 384 * 1536, lwt, 384, 1536);
    convw_kernel<<<(1536 * 384 * 9 + 255) / 256, 256, 0, stream>>>(cw + (size_t)i * 1536 * 384 * 9, cwt);

    // attention block
    layernorm_kernel<<<(unsigned)M1, 128, 0, stream>>>(x1p, ln1_g + i * 384, ln1_b + i * 384, qnb, nullptr);
    gemm_wmma_kernel<0, 1><<<dim3(g1, 6), 256, 0, stream>>>(qnb, 384, qwt, 384, nullptr, scaling, nullptr, qb);
    gemm_wmma_kernel<0, 1><<<dim3(g2, 6), 256, 0, stream>>>(x2b, 384, kwt, 384, nullptr, 1.f, nullptr, kb);
    gemm_wmma_kernel<0, 1><<<dim3(g2, 6), 256, 0, stream>>>(x2b, 384, vwt, 384, nullptr, 1.f, nullptr, vb);
    attention_kernel<<<dim3(16, 16), 256, 0, stream>>>(qb, kb, vb, ab, loss, (i == 0) ? 1 : 0);
    gemm_wmma_kernel<1, 1><<<dim3(g1, 6), 256, 0, stream>>>(ab, 384, owt, 384, nullptr, 1.f, x1p, nullptr);

    // FFN: conv(k=9) -> gelu -> linear, residual into x1p
    layernorm_kernel<<<(unsigned)M1, 128, 0, stream>>>(x1p, ln2_g + i * 384, ln2_b + i * 384, qnb, nullptr);
    gemm_wmma_kernel<2, 9><<<dim3(g1, 24), 256, 0, stream>>>(qnb, 384, cwt, 1536, cb + (size_t)i * 1536,
                                                             0.33333334f, nullptr, hb);
    gemm_wmma_kernel<1, 1><<<dim3(g1, 6), 256, 0, stream>>>(hb, 1536, lwt, 384, lb + (size_t)i * 384, 1.f,
                                                            x1p, nullptr);
  }

  layernorm_kernel<<<(unsigned)M1, 128, 0, stream>>>(x1p, lnf_g, lnf_b, nullptr, out);
  loss_kernel<<<1, 32, 0, stream>>>(loss, out + M1 * CN);
}
